// RasterizePointsXYsBlending_13537736917154
// MI455X (gfx1250) — matrix operand order repaired and checked by reading, not compiled, not run on hardware
//
#include <hip/hip_runtime.h>

typedef __attribute__((ext_vector_type(16))) _Float16 v16h;
typedef __attribute__((ext_vector_type(8)))  float    v8f;

#define HH   256
#define WW   512
#define KK   8
#define PN   200000
#define CN   64
#define HWP  (HH*WW)
#define R2F  5.4931640625e-4f   /* (3/256*2)^2 */
#define EMPTY_KEY 0xFFFFFFFFFFFFFFFFull

// ---------------- kernel 0: init workspace ----------------
__global__ void k_init(unsigned long long* kbuf, unsigned* maxbits) {
    int i = blockIdx.x * blockDim.x + threadIdx.x;
    if (i < HWP * KK) kbuf[i] = EMPTY_KEY;
    if (i == 0) *maxbits = 0u;
}

// ---------------- kernel 1: global max of point distances ----------------
__global__ void k_maxdist(const float* __restrict__ coords, unsigned* maxbits) {
    __shared__ float red[256];
    int i = blockIdx.x * blockDim.x + threadIdx.x;
    float d = 0.0f;
    if (i < PN) {
        float x = coords[i*3+0], y = coords[i*3+1], z = coords[i*3+2];
        d = sqrtf(x*x + y*y + z*z);
    }
    red[threadIdx.x] = d;
    __syncthreads();
    for (int s = 128; s > 0; s >>= 1) {
        if (threadIdx.x < s) red[threadIdx.x] = fmaxf(red[threadIdx.x], red[threadIdx.x + s]);
        __syncthreads();
    }
    if (threadIdx.x == 0) atomicMax(maxbits, __float_as_uint(red[0]));
}

// ---------------- kernel 2: project + scatter into per-pixel K-buffer ----------------
__global__ void k_scatter(const float* __restrict__ coords, const unsigned* __restrict__ maxbits,
                          float2* __restrict__ proj, unsigned long long* __restrict__ kbuf) {
    int p = blockIdx.x * blockDim.x + threadIdx.x;
    if (p >= PN) return;
    float x = coords[p*3+0], y = coords[p*3+1], z = coords[p*3+2];
    float dist = sqrtf(x*x + y*y + z*z);
    float inv  = 1.0f / dist;
    const float INV_PI = 0.318309886183790671f;
    float lat = asinf(z * inv);
    float lon = atan2f(x * inv, y * inv);
    float px  = -2.0f * lon * INV_PI;           // -x*2, x = lon/pi
    float py  =  2.0f * lat * INV_PI;           // lat/(pi/2)
    float pz  = dist / __uint_as_float(*maxbits);
    proj[p] = make_float2(px, py);

    int ixc = __float2int_rn((1.0f - px * 0.5f) * (WW * 0.5f) - 0.5f);  // ASPECT=2
    int iyc = __float2int_rn((1.0f - py)        * (HH * 0.5f) - 0.5f);

    unsigned long long key = ((unsigned long long)__float_as_uint(pz) << 32) | (unsigned)p;

    for (int dy = -3; dy <= 3; ++dy) {
        int iy = iyc + dy;
        float cy  = 1.0f - (2*iy + 1) * (1.0f / HH);
        float dy2 = (py - cy) * (py - cy);
        for (int dx = -3; dx <= 3; ++dx) {
            int ix = ixc + dx;
            float cx = 2.0f * (1.0f - (2*ix + 1) * (1.0f / WW));
            float d2 = (px - cx) * (px - cx) + dy2;
            if (d2 < R2F && iy >= 0 && iy < HH && ix >= 0 && ix < WW) {
                unsigned long long cand = key;
                unsigned long long* slot = kbuf + (size_t)(iy * WW + ix) * KK;
                #pragma unroll
                for (int s = 0; s < KK; ++s) {                    // atomicMin cascade insert
                    unsigned long long prev = atomicMin(&slot[s], cand);
                    if (prev == EMPTY_KEY) break;                 // landed in empty slot
                    cand = prev > cand ? prev : cand;             // carry displaced value
                }
            }
        }
    }
}

// ---------------- kernel 3: resolve + composite via WMMA ----------------
// One wave32 handles 16 pixels: D(16ch x 16px) = A(16x128 feats^T) * B(128x16 blockdiag weights)
__global__ __launch_bounds__(256) void k_composite(const unsigned long long* __restrict__ kbuf,
                                                   const float2* __restrict__ proj,
                                                   const float*  __restrict__ feats,
                                                   float* __restrict__ out) {
    __shared__ float s_w[8][16][KK];
    __shared__ int   s_pid[8][16][KK];
    int lane = threadIdx.x & 31;
    int wave = threadIdx.x >> 5;
    int tileBase = (blockIdx.x * 8 + wave) * 16;

    if (lane < 16) {
        int pix = tileBase + lane;
        unsigned long long k0[KK];
        const unsigned long long* kp = kbuf + (size_t)pix * KK;
        #pragma unroll
        for (int s = 0; s < KK; ++s) k0[s] = kp[s];
        // 8-input sorting network (19 CEs): ascending (z, pid)
        #define CE(a,b) { if (k0[a] > k0[b]) { unsigned long long t=k0[a]; k0[a]=k0[b]; k0[b]=t; } }
        CE(0,2) CE(1,3) CE(4,6) CE(5,7)
        CE(0,4) CE(1,5) CE(2,6) CE(3,7)
        CE(0,1) CE(2,3) CE(4,5) CE(6,7)
        CE(2,4) CE(3,5)
        CE(1,4) CE(3,6)
        CE(1,2) CE(3,4) CE(5,6)
        #undef CE
        int xI = pix & (WW - 1);
        int yI = pix >> 9;
        float cx = 2.0f * (1.0f - (2*xI + 1) * (1.0f / WW));
        float cy = 1.0f - (2*yI + 1) * (1.0f / HH);
        float T = 1.0f;
        #pragma unroll
        for (int s = 0; s < KK; ++s) {
            float w = 0.0f; int pid = 0;
            if (k0[s] != EMPTY_KEY) {
                pid = (int)(unsigned)(k0[s] & 0xFFFFFFFFull);
                float2 pp = proj[pid];
                float ddx = pp.x - cx, ddy = pp.y - cy;
                float d2  = fminf(fmaxf(ddx*ddx + ddy*ddy, 0.001f), 1.0f);
                float a   = 1.0f - sqrtf(d2);          // TAU = 1
                w  = a * T;
                T *= (1.0f - a);
            }
            s_w[wave][lane][s]   = w;
            s_pid[wave][lane][s] = pid;
        }
    }
    __syncthreads();

    bool hi = lane >= 16;
    int  lm = lane & 15;

    // B fragments: block-diagonal weight matrix, one per 32-wide K chunk (reused over channel tiles).
    // B 32x16 in this code's contiguous K order (same as A): lanes 0-15 hold K=0..15, lanes 16-31 hold K=16..31, column = lane&15.
    v16h bfrag[4];
    #pragma unroll
    for (int c = 0; c < 4; ++c) {
        #pragma unroll
        for (int j = 0; j < 16; ++j) {
            int kk   = (hi ? 16 : 0) + j;
            int s    = 32*c + kk;            // global K row = slot index over 16px*8slots
            int pxl  = s >> 3;
            int slot = s & 7;
            float v  = (pxl == lm) ? s_w[wave][lm][slot] : 0.0f;
            bfrag[c][j] = (_Float16)v;
        }
    }

    // Channel tiles: M = 16 channels, N = 16 pixels, K = 128 (4 chunks of 32).
    #pragma unroll
    for (int t = 0; t < 4; ++t) {
        v8f acc = {};
        #pragma unroll
        for (int c = 0; c < 4; ++c) {
            // A 16x32 in this code's contiguous K order (same as B): row M = lane&15; lanes 0-15 hold K 0-15, lanes 16-31 hold K 16-31
            v16h afrag;
            #pragma unroll
            for (int j = 0; j < 16; ++j) {
                int kk  = (hi ? 16 : 0) + j;
                int s   = 32*c + kk;
                int pid = s_pid[wave][s >> 3][s & 7];
                afrag[j] = (_Float16)feats[(size_t)pid * CN + 16*t + lm];
            }
            acc = __builtin_amdgcn_wmma_f32_16x16x32_f16(
                false, afrag, false, bfrag[c], (short)0, acc, false, false);
        }
        // D layout: lane holds column N=lane&15 (pixel); VGPR r holds M=r (lanes 0-15) / M=8+r (lanes 16-31)
        #pragma unroll
        for (int r = 0; r < 8; ++r) {
            int ch = 16*t + r + (hi ? 8 : 0);
            out[(size_t)ch * HWP + tileBase + lm] = acc[r];   // coalesced over pixels
        }
    }
}

extern "C" void kernel_launch(void* const* d_in, const int* in_sizes, int n_in,
                              void* d_out, int out_size, void* d_ws, size_t ws_size,
                              hipStream_t stream) {
    const float* coords = (const float*)d_in[0];
    const float* feats  = (const float*)d_in[1];
    float* out = (float*)d_out;

    char* ws = (char*)d_ws;
    unsigned* maxbits        = (unsigned*)ws;                      // 4B (padded to 256)
    float2*   proj           = (float2*)(ws + 256);                // PN * 8B = 1.6MB
    unsigned long long* kbuf = (unsigned long long*)(ws + 256 + (size_t)PN * sizeof(float2)); // 8MB

    int initN = HWP * KK;
    hipLaunchKernelGGL(k_init,    dim3((initN + 255) / 256), dim3(256), 0, stream, kbuf, maxbits);
    hipLaunchKernelGGL(k_maxdist, dim3((PN + 255) / 256),    dim3(256), 0, stream, coords, maxbits);
    hipLaunchKernelGGL(k_scatter, dim3((PN + 255) / 256),    dim3(256), 0, stream, coords, maxbits, proj, kbuf);
    hipLaunchKernelGGL(k_composite, dim3(HWP / 128),         dim3(256), 0, stream, kbuf, proj, feats, out);
}
